// LSTMModel_25993142075827
// MI455X (gfx1250) — matrix operand order repaired and checked by reading, not compiled, not run on hardware
//
#include <hip/hip_runtime.h>

#define BB 8192
#define TT 512
#define II 3
#define HH 25
#define NG 100     // 4*H gates
#define NT 7       // gate tiles of 16 (100 -> 112)
#define GS 112     // padded gate stride in LDS
#define KK 32      // augmented/padded K: [h(25) | x(3) | 1 | 0,0,0]

typedef _Float16 half_t;
typedef __attribute__((ext_vector_type(16))) _Float16 v16h;
typedef __attribute__((ext_vector_type(8)))  float    v8f;
typedef __attribute__((ext_vector_type(2)))  _Float16 h2;

// TRANS-unit activations: v_exp_f32 + v_rcp_f32 (co-execute with XDL WMMA)
__device__ __forceinline__ float fast_sigmoid(float v) {
    return __builtin_amdgcn_rcpf(1.0f + __expf(-v));
}
__device__ __forceinline__ float fast_tanh(float v) {
    float e = __expf(2.0f * v);
    return 1.0f - 2.0f * __builtin_amdgcn_rcpf(e + 1.0f);
}

// One wave (32 threads) owns 16 batch rows. Per time step:
//   gates^T[100x16] = A[100x32] x B[32x16]  via 7x v_wmma_f32_16x16x32_f16
// A = [w_hh | w_ih | (b_ih+b_hh) | 0] (constant, preloaded in VGPRs, f16)
// B = [h ; x_t ; 1 ; 0] per batch column (rebuilt from LDS each step, f16)
__global__ __launch_bounds__(32)
void lstm_wmma_kernel(const float* __restrict__ x,
                      const float* __restrict__ w_ih,
                      const float* __restrict__ w_hh,
                      const float* __restrict__ b_ih,
                      const float* __restrict__ b_hh,
                      const float* __restrict__ w_lin,
                      const float* __restrict__ b_lin,
                      float* __restrict__ out) {
    __shared__ float   gbuf[16 * GS];   // gates [batch][gate], f32 (7168 B)
    __shared__ half_t  hbuf[16 * KK];   // h (f16) [batch][k], k<25 valid (1024 B)
    __shared__ float   acc[16];

    const int  lane    = threadIdx.x;
    const int  half_id = lane >> 4;     // 0: K 0..15 side, 1: K 16..31 side
    const bool hi      = half_id != 0;
    const int  lm      = lane & 15;
    const int  b0      = blockIdx.x * 16;

    // h0 = 0
    {
        unsigned int* hz = (unsigned int*)hbuf;
        for (int i = lane; i < (16 * KK) / 2; i += 32) hz[i] = 0u;
    }
    if (lane < 16) acc[lane] = 0.0f;

    // ---- Preload augmented weights into 7 A fragments (f16 16x32 layout):
    // lane<16 : element e -> K = e        (K 0..15)
    // lane>=16: element e -> K = e + 16   (K 16..31)
    v16h a[NT];
    #pragma unroll
    for (int t = 0; t < NT; ++t) {
        const int m = 16 * t + lm;  // gate row
        #pragma unroll
        for (int e = 0; e < 16; ++e) {
            const int k = e + (half_id ? 16 : 0);
            float av = 0.0f;
            if (m < NG) {
                if (k < HH)            av = w_hh[m * HH + k];
                else if (k < HH + II)  av = w_ih[m * II + (k - HH)];
                else if (k == HH + II) av = b_ih[m] + b_hh[m];  // bias row, x1.0
            }
            a[t][e] = (half_t)av;
        }
    }

    // lane owns batch lm, hidden units j = 2q + half_id.
    // q=12 on odd half is a padded slot (j=25): all its reads/writes are
    // in-bounds, its gates read rows 25/50/75 (real) and 100 (zero pad), and its h is
    // never consumed (B element 9 is overwritten by x) -> no guards needed.
    float creg[13];
    float hreg[13];
    #pragma unroll
    for (int q = 0; q < 13; ++q) { creg[q] = 0.0f; hreg[q] = 0.0f; }

    const float* xrow = x + (size_t)(b0 + lm) * TT * II;

    __syncthreads();

    for (int step = 0; step < TT; ++step) {
        // x_t loaded by ALL lanes (L and L+16 share a batch -> same address),
        // so the load issues unpredicated and early.
        const float* xp = xrow + (size_t)step * II;
        const float x0 = xp[0], x1 = xp[1], x2 = xp[2];
        __builtin_prefetch(xp + II, 0, 0);

        // ---- Build B fragment column [h ; x_t ; 1 ; 0] for batch lm ----
        v16h bm;
        const unsigned int* hb =
            (const unsigned int*)(hbuf + (lm * KK + half_id * 16));
        #pragma unroll
        for (int d = 0; d < 8; ++d) {
            h2 p = __builtin_bit_cast(h2, hb[d]);
            bm[2 * d]     = p.x;
            bm[2 * d + 1] = p.y;
        }
        // High half: elements 8..15 = K 24..31: h[24], x0,x1,x2, 1, 0,0,0.
        // Branchless per-element selects (v_cndmask) instead of saveexec.
        bm[9]  = hi ? (half_t)x0   : bm[9];
        bm[10] = hi ? (half_t)x1   : bm[10];
        bm[11] = hi ? (half_t)x2   : bm[11];
        bm[12] = hi ? (half_t)1.0f : bm[12];
        bm[13] = hi ? (half_t)0.0f : bm[13];
        bm[14] = hi ? (half_t)0.0f : bm[14];
        bm[15] = hi ? (half_t)0.0f : bm[15];

        // ---- 7 WMMAs: gate tile t -> rows 16t..16t+15, cols = 16 batches ----
        #pragma unroll
        for (int t = 0; t < NT; ++t) {
            v8f c = {};
            v8f d = __builtin_amdgcn_wmma_f32_16x16x32_f16(
                false, a[t], false, bm, (short)0, c, false, false);
            // lane holds 8 consecutive gate rows for batch column lm
            float* gp = &gbuf[lm * GS + 16 * t + half_id * 8];
            #pragma unroll
            for (int r = 0; r < 8; ++r) gp[r] = d[r];
        }
        __syncthreads();  // single-wave WG: lowers to waits/S_NOP

        // ---- Elementwise LSTM cell update (gate order i,f,g,o), branchless --
        const float* grow = &gbuf[lm * GS];
        #pragma unroll
        for (int q = 0; q < 13; ++q) {
            const int j = 2 * q + half_id;
            const float gi = grow[j];
            const float gf = grow[HH + j];
            const float gg = grow[2 * HH + j];
            const float go = grow[3 * HH + j];
            const float cn = fast_sigmoid(gf) * creg[q] +
                             fast_sigmoid(gi) * fast_tanh(gg);
            creg[q] = cn;
            const float hn = fast_sigmoid(go) * fast_tanh(cn);
            hreg[q] = hn;
            hbuf[lm * KK + j] = (half_t)hn;   // f16 h for next step's B
        }
        __syncthreads();
    }

    // ---- Final linear: out[b] = sum_j h[b][j] * w_lin[j] + b_lin ----
    float part = 0.0f;
    #pragma unroll
    for (int q = 0; q < 13; ++q) {
        const int j = 2 * q + half_id;
        const float wl = (j < HH) ? w_lin[j] : 0.0f;  // mask padded slot
        part += hreg[q] * wl;
    }
    atomicAdd(&acc[lm], part);   // ds_add_f32, two lanes per batch
    __syncthreads();
    if (lane < 16) out[b0 + lane] = acc[lane] + b_lin[0];
}

extern "C" void kernel_launch(void* const* d_in, const int* in_sizes, int n_in,
                              void* d_out, int out_size, void* d_ws, size_t ws_size,
                              hipStream_t stream) {
    const float* x     = (const float*)d_in[0];
    const float* w_ih  = (const float*)d_in[1];
    const float* w_hh  = (const float*)d_in[2];
    const float* b_ih  = (const float*)d_in[3];
    const float* b_hh  = (const float*)d_in[4];
    const float* w_lin = (const float*)d_in[5];
    const float* b_lin = (const float*)d_in[6];
    float* out = (float*)d_out;

    dim3 grid(BB / 16);   // 512 waves, one 16-batch tile each
    dim3 block(32);       // one wave32 per workgroup
    lstm_wmma_kernel<<<grid, block, 0, stream>>>(x, w_ih, w_hh, b_ih, b_hh,
                                                 w_lin, b_lin, out);
}
